// GASP_EGNN_19748259627798
// MI455X (gfx1250) — matrix operand order repaired and checked by reading, not compiled, not run on hardware
//
#include <hip/hip_runtime.h>
#include <hip/hip_bf16.h>
#include <math.h>

// ---- CDNA5 vector types for WMMA ----
typedef __fp16   f16x2 __attribute__((ext_vector_type(2)));
typedef _Float16 v8h  __attribute__((ext_vector_type(8)));
typedef _Float16 v16h __attribute__((ext_vector_type(16)));
typedef float    v8f  __attribute__((ext_vector_type(8)));

#define HDIM    64
#define EIN     8
#define KPAD    160          // 2*H + EIN = 136, padded to 5 * 32
#define ASTRIDE 176          // A-panel row stride in halves (16B-aligned rows)
#define XSTRIDE 80           // x1 panel row stride in halves
#define WAVES   4
#define BLOCK   (WAVES * 32)

__global__ __launch_bounds__(256) void egnn_init(const float* __restrict__ coord,
                                                 float* __restrict__ out, int n) {
  int i = blockIdx.x * blockDim.x + threadIdx.x;
  if (i < n) out[i] = coord[i];
}

__device__ __forceinline__ float silu(float x) {
  return x / (1.0f + __expf(-x));
}

// two f32 -> one packed dword of two f16 via v_cvt_pk_rtz_f16_f32
__device__ __forceinline__ unsigned int pk(float x, float y) {
  f16x2 p = __builtin_amdgcn_cvt_pkrtz(x, y);
  return __builtin_bit_cast(unsigned int, p);
}

// 8 f32 -> 4 dwords of packed f16
__device__ __forceinline__ uint4 pack8(float4 a, float4 b) {
  uint4 o;
  o.x = pk(a.x, a.y);
  o.y = pk(a.z, a.w);
  o.z = pk(b.x, b.y);
  o.w = pk(b.z, b.w);
  return o;
}

__global__ __launch_bounds__(BLOCK) void egnn_edge(
    const float* __restrict__ h, const float* __restrict__ coord,
    const int* __restrict__ ei, const float* __restrict__ ea,
    const float* __restrict__ W1, const float* __restrict__ b1,
    const float* __restrict__ W2, const float* __restrict__ b2,
    const float* __restrict__ W3, float* __restrict__ out,
    int E, int tiles) {
  // LDS: ~62.5 KB total
  __shared__ _Float16 sW1[64 * KPAD];            // W1^T  [n][k], k padded with 0
  __shared__ _Float16 sW2[64 * 64];              // W2^T  [n][k]
  __shared__ float    sB1[64], sB2[64], sW3[64];
  __shared__ _Float16 sA[WAVES * 16 * ASTRIDE];  // per-wave input panel [m][k]
  __shared__ _Float16 sX[WAVES * 16 * XSTRIDE];  // per-wave x1 panel    [m][k]
  __shared__ float    sScal[WAVES * 16];

  const int tid = threadIdx.x;

  // ---- stage weights (f32 -> f16, transposed) ----
  for (int idx = tid; idx < 64 * 136; idx += BLOCK) {
    int k = idx >> 6, n = idx & 63;             // coalesced read of W1[k][n]
    sW1[n * KPAD + k] = (_Float16)W1[idx];
  }
  for (int idx = tid; idx < 64 * 24; idx += BLOCK) {
    int n = idx & 63, k = 136 + (idx >> 6);     // zero K-padding
    sW1[n * KPAD + k] = (_Float16)0.0f;
  }
  for (int idx = tid; idx < 64 * 64; idx += BLOCK) {
    int k = idx >> 6, n = idx & 63;
    sW2[n * 64 + k] = (_Float16)W2[idx];
  }
  if (tid < 64) { sB1[tid] = b1[tid]; sB2[tid] = b2[tid]; sW3[tid] = W3[tid]; }
  __syncthreads();

  const int wave = tid >> 5;
  const int lane = tid & 31;
  const int mrow = lane & 15;   // matrix row (edge within tile) / column index
  const int half = lane >> 4;   // lane half selects K sub-range per ISA layouts
  _Float16* wA = &sA[wave * 16 * ASTRIDE];
  _Float16* wX = &sX[wave * 16 * XSTRIDE];
  float*    wS = &sScal[wave * 16];

  for (int t = blockIdx.x; t < tiles; t += gridDim.x) {
    const int base = t * (WAVES * 16) + wave * 16;
    const int e  = base + mrow;
    const int ec = (e < E) ? e : (E - 1);
    const int r  = ei[ec];
    const int c  = ei[E + ec];

    // ---- stage A panel: [h[row] | h[col] | edge_attr | 0-pad] as f16 ----
    {
      const float* hr = h + (size_t)r * HDIM + half * 32;
      const float* hc = h + (size_t)c * HDIM + half * 32;
      _Float16* dst = wA + mrow * ASTRIDE + half * 32;
      #pragma unroll
      for (int i = 0; i < 32; i += 8) {
        float4 va = *(const float4*)(hr + i);
        float4 vb = *(const float4*)(hr + i + 4);
        *(uint4*)(dst + i) = pack8(va, vb);
      }
      #pragma unroll
      for (int i = 0; i < 32; i += 8) {
        float4 va = *(const float4*)(hc + i);
        float4 vb = *(const float4*)(hc + i + 4);
        *(uint4*)(dst + 64 + i) = pack8(va, vb);
      }
      if (half == 0) {
        float4 ea0 = *(const float4*)(ea + (size_t)ec * EIN);
        float4 ea1 = *(const float4*)(ea + (size_t)ec * EIN + 4);
        *(uint4*)(wA + mrow * ASTRIDE + 128) = pack8(ea0, ea1);
        uint4 z = {0u, 0u, 0u, 0u};                 // pad 136..143
        *(uint4*)(wA + mrow * ASTRIDE + 136) = z;
      } else {
        uint4 z = {0u, 0u, 0u, 0u};                 // pad 144..159
        *(uint4*)(wA + mrow * ASTRIDE + 144) = z;
        *(uint4*)(wA + mrow * ASTRIDE + 152) = z;
      }
    }

    // ---- software prefetch of NEXT tile's gather rows into L2/WGP$ ----
    {
      const int tn = t + (int)gridDim.x;
      if (tn < tiles) {
        int en = tn * (WAVES * 16) + wave * 16 + mrow;
        if (en < E) {
          int rn = ei[en], cn = ei[E + en];
          __builtin_prefetch(h + (size_t)rn * HDIM + half * 32, 0, 1);
          __builtin_prefetch(h + (size_t)cn * HDIM + half * 32, 0, 1);
          __builtin_prefetch(ea + (size_t)en * EIN, 0, 1);
        }
      }
    }
    asm volatile("s_wait_dscnt 0" ::: "memory");  // intra-wave panel visibility

    // ---- layer 1: [16 x 160] @ [160 x 64] via 5x4 WMMAs ----
    v8f acc[4];
    #pragma unroll
    for (int ct = 0; ct < 4; ++ct) {
      v8f z = {0.f, 0.f, 0.f, 0.f, 0.f, 0.f, 0.f, 0.f};
      acc[ct] = z;
    }
    #pragma unroll
    for (int kc = 0; kc < 5; ++kc) {
      // A frag (16-bit A layout): lane holds K = kc*32 + half*8 .. +8 and +16..
      const _Float16* ap = wA + mrow * ASTRIDE + kc * 32 + half * 8;
      v8h alo = *(const v8h*)ap;
      v8h ahi = *(const v8h*)(ap + 16);
      v16h a;
      #pragma unroll
      for (int i = 0; i < 8; ++i) { a[i] = alo[i]; a[i + 8] = ahi[i]; }
      #pragma unroll
      for (int ct = 0; ct < 4; ++ct) {
        // B frag: lane = column n, K = kc*32 + half*8 .. +8 and +16.. (matches A)
        const _Float16* bp = sW1 + (ct * 16 + mrow) * KPAD + kc * 32 + half * 8;
        v8h blo = *(const v8h*)bp;
        v8h bhi = *(const v8h*)(bp + 16);
        v16h b;
        #pragma unroll
        for (int i = 0; i < 8; ++i) { b[i] = blo[i]; b[i + 8] = bhi[i]; }
        acc[ct] = __builtin_amdgcn_wmma_f32_16x16x32_f16(
            false, a, false, b, (short)0, acc[ct], false, false);
      }
    }

    // ---- bias + SiLU, transpose D-layout back to [m][k] f16 panel ----
    #pragma unroll
    for (int ct = 0; ct < 4; ++ct) {
      const float bb = sB1[ct * 16 + mrow];        // D: column n = lane&15
      #pragma unroll
      for (int j = 0; j < 8; ++j) {                // D: row m = half*8 + j
        float x = acc[ct][j] + bb;
        wX[(half * 8 + j) * XSTRIDE + ct * 16 + mrow] = (_Float16)silu(x);
      }
    }
    asm volatile("s_wait_dscnt 0" ::: "memory");

    // ---- layer 2: [16 x 64] @ [64 x 64] via 2x4 WMMAs ----
    v8f acc2[4];
    #pragma unroll
    for (int ct = 0; ct < 4; ++ct) {
      v8f z = {0.f, 0.f, 0.f, 0.f, 0.f, 0.f, 0.f, 0.f};
      acc2[ct] = z;
    }
    #pragma unroll
    for (int kc = 0; kc < 2; ++kc) {
      const _Float16* ap = wX + mrow * XSTRIDE + kc * 32 + half * 8;
      v8h alo = *(const v8h*)ap;
      v8h ahi = *(const v8h*)(ap + 16);
      v16h a;
      #pragma unroll
      for (int i = 0; i < 8; ++i) { a[i] = alo[i]; a[i + 8] = ahi[i]; }
      #pragma unroll
      for (int ct = 0; ct < 4; ++ct) {
        const _Float16* bp = sW2 + (ct * 16 + mrow) * 64 + kc * 32 + half * 8;
        v8h blo = *(const v8h*)bp;
        v8h bhi = *(const v8h*)(bp + 16);
        v16h b;
        #pragma unroll
        for (int i = 0; i < 8; ++i) { b[i] = blo[i]; b[i + 8] = bhi[i]; }
        acc2[ct] = __builtin_amdgcn_wmma_f32_16x16x32_f16(
            false, a, false, b, (short)0, acc2[ct], false, false);
      }
    }

    // ---- bias + SiLU + 64->1 projection: per-lane partials over n=lane&15 ----
    float part[8];
    #pragma unroll
    for (int j = 0; j < 8; ++j) part[j] = 0.0f;
    #pragma unroll
    for (int ct = 0; ct < 4; ++ct) {
      const float bb = sB2[ct * 16 + mrow];
      const float w3 = sW3[ct * 16 + mrow];
      #pragma unroll
      for (int j = 0; j < 8; ++j) {
        float x = acc2[ct][j] + bb;
        part[j] += silu(x) * w3;
      }
    }
    // reduce over the 16 lanes of each half (columns n) -> scal per edge row
    #pragma unroll
    for (int j = 0; j < 8; ++j) {
      float v = part[j];
      v += __shfl_xor(v, 1, 32);
      v += __shfl_xor(v, 2, 32);
      v += __shfl_xor(v, 4, 32);
      v += __shfl_xor(v, 8, 32);
      part[j] = v;
    }
    if (mrow == 0) {
      #pragma unroll
      for (int j = 0; j < 8; ++j) wS[half * 8 + j] = part[j];
    }
    asm volatile("s_wait_dscnt 0" ::: "memory");

    // ---- coordinate update + scatter (lanes 0..15 own edge m = mrow) ----
    if (half == 0 && e < E) {
      const float scal = wS[mrow];
      const float* pr = coord + (size_t)r * 3;
      const float* pc = coord + (size_t)c * 3;
      float dx = pr[0] - pc[0];
      float dy = pr[1] - pc[1];
      float dz = pr[2] - pc[2];
      float radial = dx * dx + dy * dy + dz * dz;
      float nrm = sqrtf(radial + 1e-8f);
      float f = scal / (nrm + 1.0f) * 0.01f;   // / (norm+NORM_CONSTANT) / NORM_FACTOR
      atomicAdd(&out[(size_t)r * 3 + 0], dx * f);
      atomicAdd(&out[(size_t)r * 3 + 1], dy * f);
      atomicAdd(&out[(size_t)r * 3 + 2], dz * f);
    }
  }
}

extern "C" void kernel_launch(void* const* d_in, const int* in_sizes, int n_in,
                              void* d_out, int out_size, void* d_ws, size_t ws_size,
                              hipStream_t stream) {
  const float* h     = (const float*)d_in[0];
  const float* coord = (const float*)d_in[1];
  const int*   ei    = (const int*)d_in[2];
  const float* ea    = (const float*)d_in[3];
  const float* W1    = (const float*)d_in[4];
  const float* b1    = (const float*)d_in[5];
  const float* W2    = (const float*)d_in[6];
  const float* b2    = (const float*)d_in[7];
  const float* W3    = (const float*)d_in[8];
  float* out = (float*)d_out;

  const int E = in_sizes[2] / 2;
  const int tiles = (E + (WAVES * 16) - 1) / (WAVES * 16);
  const int grid = tiles < 2048 ? tiles : 2048;

  // out = coord (re-seeded every launch; harness poisons d_out)
  egnn_init<<<(out_size + 255) / 256, 256, 0, stream>>>(coord, out, out_size);
  // out += segment_sum(trans) / NORM_FACTOR via f32 global atomics
  egnn_edge<<<grid, BLOCK, 0, stream>>>(h, coord, ei, ea, W1, b1, W2, b2, W3,
                                        out, E, tiles);
}
